// AttentionBlock_68624987455817
// MI455X (gfx1250) — matrix operand order repaired and checked by reading, not compiled, not run on hardware
//
#include <hip/hip_runtime.h>
#include <hip/hip_bf16.h>
#include <cstdint>
#include <cstddef>

// ---------- types ----------
typedef __attribute__((ext_vector_type(16))) __bf16       v16bf;
typedef __attribute__((ext_vector_type(8)))  float        v8f;
typedef __attribute__((ext_vector_type(4)))  unsigned int u32x4;
typedef __attribute__((ext_vector_type(4)))  float        f32x4;

union ABFrag { u32x4 q[2]; v16bf v; };   // 32 bytes: one WMMA A/B fragment (16 bf16/lane)

__device__ __forceinline__ unsigned short f2bf(float f) {
  unsigned int u = __float_as_uint(f);
  u += 0x7FFFu + ((u >> 16) & 1u);       // round-to-nearest-even
  return (unsigned short)(u >> 16);
}

__device__ __forceinline__ v8f vzero8() {
  v8f r;
#pragma unroll
  for (int i = 0; i < 8; ++i) r[i] = 0.0f;
  return r;
}

// CDNA5 async global->LDS copy (ASYNCcnt). LDS operand = per-lane LDS byte
// offset (== low 32 bits of the generic pointer into __shared__). INST_OFFSET
// is added to BOTH the LDS and global addresses (ISA 10. async pseudocode),
// so one base pair covers consecutive 16B chunks.
__device__ __forceinline__ void async_copy32(unsigned int ldsOff, const void* g) {
  asm volatile(
      "global_load_async_to_lds_b128 %0, %1, off\n\t"
      "global_load_async_to_lds_b128 %0, %1, off offset:16"
      :: "v"(ldsOff), "v"((unsigned long long)(size_t)g) : "memory");
}
__device__ __forceinline__ void async_copy64(unsigned int ldsOff, const void* g) {
  asm volatile(
      "global_load_async_to_lds_b128 %0, %1, off\n\t"
      "global_load_async_to_lds_b128 %0, %1, off offset:16\n\t"
      "global_load_async_to_lds_b128 %0, %1, off offset:32\n\t"
      "global_load_async_to_lds_b128 %0, %1, off offset:48"
      :: "v"(ldsOff), "v"((unsigned long long)(size_t)g) : "memory");
}
__device__ __forceinline__ void wait_async0() {
  asm volatile("s_wait_asynccnt 0" ::: "memory");
}

// ---------- LayerNorm -> bf16 ----------
__global__ __launch_bounds__(256) void ln_bf16_kernel(
    const float* __restrict__ x, const float* __restrict__ scale,
    const float* __restrict__ bias, unsigned short* __restrict__ z) {
  constexpr int E = 768;
  __shared__ float redS[8], redQ[8];
  const int row = blockIdx.x;
  const int tid = threadIdx.x;
  const int lane = tid & 31;
  const int w = tid >> 5;
  float v[3];
  float s = 0.f, q = 0.f;
#pragma unroll
  for (int k = 0; k < 3; ++k) {
    v[k] = x[(size_t)row * E + tid + k * 256];
    s += v[k];
    q += v[k] * v[k];
  }
#pragma unroll
  for (int off = 16; off >= 1; off >>= 1) {
    s += __shfl_xor(s, off, 32);
    q += __shfl_xor(q, off, 32);
  }
  if (lane == 0) { redS[w] = s; redQ[w] = q; }
  __syncthreads();
  float ts = 0.f, tq = 0.f;
#pragma unroll
  for (int i = 0; i < 8; ++i) { ts += redS[i]; tq += redQ[i]; }
  const float mean = ts * (1.0f / E);
  const float var  = tq * (1.0f / E) - mean * mean;
  const float rstd = rsqrtf(var + 1e-6f);
#pragma unroll
  for (int k = 0; k < 3; ++k) {
    const int c = tid + k * 256;
    z[(size_t)row * E + c] = f2bf((v[k] - mean) * rstd * scale[c] + bias[c]);
  }
}

// ---------- GEMM: out = epi(A_bf16[MxK] @ W_f32[KxN] + bias) ----------
// EPI 0: +bias            -> bf16
// EPI 1: +bias +resid_f32 -> f32
// EPI 2: +bias, SELU      -> bf16
template <int EPI>
__global__ __launch_bounds__(256) void gemm_bf16_kernel(
    const unsigned short* __restrict__ A, const float* __restrict__ W,
    const float* __restrict__ bias, const float* __restrict__ resid,
    void* __restrict__ out, int M, int N, int K) {
  __shared__ __align__(16) unsigned short ldsA[2][128 * 40];  // 128 rows x 32 K (+8 pad)
  __shared__ __align__(16) unsigned short ldsB[2][128 * 40];  // 128 cols x 32 K (+8 pad), N-major

  const int tid  = threadIdx.x;
  const int lane = tid & 31;
  const int w    = tid >> 5;
  const int lr   = lane & 15;
  const int half = lane >> 4;
  const int wm   = w & 3;    // 4 waves along M (32 rows each)
  const int wn   = w >> 2;   // 2 waves along N (64 cols each)
  const int m0   = blockIdx.y * 128;
  const int n0   = blockIdx.x * 128;

  v8f acc[2][4];
#pragma unroll
  for (int mi = 0; mi < 2; ++mi)
#pragma unroll
    for (int ni = 0; ni < 4; ++ni) acc[mi][ni] = vzero8();

  const int nk = K / 32;

#define GEMM_STAGE(BUF, K0)                                                        \
  do {                                                                             \
    {   /* A tile: bf16 verbatim -> async DMA into LDS */                          \
      const int ar = tid >> 1;                                                     \
      const int ac = (tid & 1) * 16;                                               \
      async_copy32((unsigned int)(size_t)&ldsA[BUF][ar * 40 + ac],                 \
                   A + (size_t)(m0 + ar) * K + (K0) + ac);                         \
    }                                                                              \
    {   /* W tile: fp32 -> bf16 convert, store N-major (transposed) */             \
      const int br = tid >> 3;                                                     \
      const int bc = (tid & 7) * 16;                                               \
      const f32x4* src = (const f32x4*)(W + (size_t)((K0) + br) * N + n0 + bc);    \
      _Pragma("unroll") for (int v4 = 0; v4 < 4; ++v4) {                           \
        f32x4 f = src[v4];                                                         \
        const int c = bc + v4 * 4;                                                 \
        ldsB[BUF][(c + 0) * 40 + br] = f2bf(f[0]);                                 \
        ldsB[BUF][(c + 1) * 40 + br] = f2bf(f[1]);                                 \
        ldsB[BUF][(c + 2) * 40 + br] = f2bf(f[2]);                                 \
        ldsB[BUF][(c + 3) * 40 + br] = f2bf(f[3]);                                 \
      }                                                                            \
    }                                                                              \
  } while (0)

  GEMM_STAGE(0, 0);
  wait_async0();
  __syncthreads();

  for (int kt = 0; kt < nk; ++kt) {
    const int cur = kt & 1;
    if (kt + 1 < nk) GEMM_STAGE(cur ^ 1, (kt + 1) * 32);  // async A overlaps WMMAs below

    ABFrag af[2], bf[4];
#pragma unroll
    for (int mi = 0; mi < 2; ++mi) {
      const u32x4* pa =
          (const u32x4*)(&ldsA[cur][(wm * 32 + mi * 16 + lr) * 40 + half * 8]);
      af[mi].q[0] = pa[0];   // K 0..7   (half0) / 8..15  (half1)
      af[mi].q[1] = pa[2];   // K 16..23 (half0) / 24..31 (half1)
    }
#pragma unroll
    for (int ni = 0; ni < 4; ++ni) {
      const u32x4* pb =
          (const u32x4*)(&ldsB[cur][(wn * 64 + ni * 16 + lr) * 40 + half * 8]);
      bf[ni].q[0] = pb[0];   // K 0..7   (half0) / 8..15  (half1)
      bf[ni].q[1] = pb[2];   // K 16..23 (half0) / 24..31 (half1)
    }
#pragma unroll
    for (int mi = 0; mi < 2; ++mi)
#pragma unroll
      for (int ni = 0; ni < 4; ++ni)
        acc[mi][ni] = __builtin_amdgcn_wmma_f32_16x16x32_bf16(
            false, af[mi].v, false, bf[ni].v, (short)0, acc[mi][ni], false, false);
    wait_async0();        // next tile's async A must land before publish barrier
    __syncthreads();
  }

  // epilogue
#pragma unroll
  for (int mi = 0; mi < 2; ++mi) {
#pragma unroll
    for (int ni = 0; ni < 4; ++ni) {
      const int gcol = n0 + wn * 64 + ni * 16 + lr;
      const float bv = bias[gcol];
#pragma unroll
      for (int j = 0; j < 8; ++j) {
        const int grow = m0 + wm * 32 + mi * 16 + half * 8 + j;
        float v = acc[mi][ni][j] + bv;
        const size_t oidx = (size_t)grow * N + gcol;
        if (EPI == 1) {
          v += resid[oidx];
          ((float*)out)[oidx] = v;
        } else if (EPI == 2) {
          v = (v > 0.0f) ? 1.0507009873554805f * v
                         : 1.7580993408473766f * (__expf(v) - 1.0f);
          ((unsigned short*)out)[oidx] = f2bf(v);
        } else {
          ((unsigned short*)out)[oidx] = f2bf(v);
        }
      }
    }
  }
#undef GEMM_STAGE
}

// ---------- flash attention ----------
// grid: (L/128, H, B); block 256 = 8 waves; wave w handles 16 query rows.
__global__ __launch_bounds__(256) void attn_kernel(
    const unsigned short* __restrict__ qkv, unsigned short* __restrict__ out) {
  constexpr int L = 1024, QKV = 2304, E = 768;
  __shared__ __align__(16) unsigned short ldsKV[9216];     // K:128x(64+8) or Vt:64x(128+8)
  __shared__ __align__(16) unsigned short ldsP[8][16 * 136];

  const int tid  = threadIdx.x;
  const int lane = tid & 31;
  const int w    = tid >> 5;
  const int lr   = lane & 15;
  const int half = lane >> 4;
  const int qt = blockIdx.x, h = blockIdx.y, b = blockIdx.z;
  const int q0 = qt * 128;

  // Q fragments (16 rows x 64 dims = 2 K-steps), read straight from global bf16
  ABFrag qf[2];
  {
    const int m = q0 + w * 16 + lr;
    const unsigned short* qbase = qkv + ((size_t)(b * L + m)) * QKV + h * 192;
#pragma unroll
    for (int s = 0; s < 2; ++s) {
      const u32x4* p = (const u32x4*)(qbase + s * 32 + half * 8);
      qf[s].q[0] = p[0];
      qf[s].q[1] = p[2];
    }
  }

  v8f o[4];
#pragma unroll
  for (int n = 0; n < 4; ++n) o[n] = vzero8();
  float mrun[8], lrun[8];
#pragma unroll
  for (int j = 0; j < 8; ++j) { mrun[j] = -3.0e38f; lrun[j] = 0.0f; }

  for (int kt = 0; kt <= qt; ++kt) {
    __syncthreads();
    {   // stage K tile: 128 keys x 64 dims, row-major — async DMA into LDS
      const int key = tid >> 1;
      const int c0  = (tid & 1) * 32;
      async_copy64((unsigned int)(size_t)&ldsKV[key * 72 + c0],
                   qkv + ((size_t)(b * L + kt * 128 + key)) * QKV + h * 192 + 64 + c0);
    }
    wait_async0();
    __syncthreads();

    // S = Q @ K^T : 16 x 128 per wave (8 N-tiles), K-dim = 64 (2 steps)
    v8f sv[8];
#pragma unroll
    for (int ni = 0; ni < 8; ++ni) sv[ni] = vzero8();
#pragma unroll
    for (int s = 0; s < 2; ++s) {
#pragma unroll
      for (int ni = 0; ni < 8; ++ni) {
        ABFrag kf;
        const u32x4* pb =
            (const u32x4*)(&ldsKV[(ni * 16 + lr) * 72 + s * 32 + half * 8]);
        kf.q[0] = pb[0];
        kf.q[1] = pb[2];
        sv[ni] = __builtin_amdgcn_wmma_f32_16x16x32_bf16(
            false, qf[s].v, false, kf.v, (short)0, sv[ni], false, false);
      }
    }
    // scale (1/sqrt(64)) + causal mask on the diagonal tile
#pragma unroll
    for (int ni = 0; ni < 8; ++ni)
#pragma unroll
      for (int j = 0; j < 8; ++j) {
        float v = sv[ni][j] * 0.125f;
        if (kt == qt) {
          const int col = kt * 128 + ni * 16 + lr;
          const int row = q0 + w * 16 + half * 8 + j;
          if (col > row) v = -1.0e30f;
        }
        sv[ni][j] = v;
      }
    // online softmax; row r lives across one 16-lane half in register j
#pragma unroll
    for (int j = 0; j < 8; ++j) {
      float mx = -1.0e30f;
#pragma unroll
      for (int ni = 0; ni < 8; ++ni) mx = fmaxf(mx, sv[ni][j]);
#pragma unroll
      for (int off = 8; off >= 1; off >>= 1) mx = fmaxf(mx, __shfl_xor(mx, off, 32));
      const float mnew = fmaxf(mrun[j], mx);
      const float corr = __expf(mrun[j] - mnew);
      float rs = 0.f;
#pragma unroll
      for (int ni = 0; ni < 8; ++ni) {
        const float p = __expf(sv[ni][j] - mnew);
        sv[ni][j] = p;
        rs += p;
      }
#pragma unroll
      for (int off = 8; off >= 1; off >>= 1) rs += __shfl_xor(rs, off, 32);
      lrun[j] = lrun[j] * corr + rs;
      mrun[j] = mnew;
#pragma unroll
      for (int n = 0; n < 4; ++n) o[n][j] *= corr;
      // C-layout -> A-layout via wave-private LDS
#pragma unroll
      for (int ni = 0; ni < 8; ++ni)
        ldsP[w][(half * 8 + j) * 136 + ni * 16 + lr] = f2bf(sv[ni][j]);
    }
    __syncthreads();
    {   // stage V tile transposed: Vt[d][key] (register path: needs transpose)
      const int key = tid >> 1;
      const int c0  = (tid & 1) * 32;
      const u32x4* src = (const u32x4*)(
          qkv + ((size_t)(b * L + kt * 128 + key)) * QKV + h * 192 + 128 + c0);
#pragma unroll
      for (int i = 0; i < 4; ++i) {
        union { u32x4 u; unsigned short hh[8]; } t;
        t.u = src[i];
#pragma unroll
        for (int e = 0; e < 8; ++e)
          ldsKV[(c0 + i * 8 + e) * 136 + key] = t.hh[e];
      }
    }
    __syncthreads();
    // O += P @ V : K-dim = 128 keys (4 steps), N = 64 dims (4 tiles)
#pragma unroll
    for (int s = 0; s < 4; ++s) {
      ABFrag pf;
      const u32x4* pp = (const u32x4*)(&ldsP[w][lr * 136 + s * 32 + half * 8]);
      pf.q[0] = pp[0];
      pf.q[1] = pp[2];
#pragma unroll
      for (int n = 0; n < 4; ++n) {
        ABFrag vf;
        const u32x4* pv =
            (const u32x4*)(&ldsKV[(n * 16 + lr) * 136 + s * 32 + half * 8]);
        vf.q[0] = pv[0];
        vf.q[1] = pv[2];
        o[n] = __builtin_amdgcn_wmma_f32_16x16x32_bf16(
            false, pf.v, false, vf.v, (short)0, o[n], false, false);
      }
    }
  }
  // normalize + write merged heads: out[b,l, h*64+d] bf16
#pragma unroll
  for (int n = 0; n < 4; ++n)
#pragma unroll
    for (int j = 0; j < 8; ++j) {
      const int row = q0 + w * 16 + half * 8 + j;
      const int d   = n * 16 + lr;
      const float v = o[n][j] / lrun[j];
      out[((size_t)(b * L + row)) * E + h * 64 + d] = f2bf(v);
    }
}

// ---------- host launch ----------
extern "C" void kernel_launch(void* const* d_in, const int* in_sizes, int n_in,
                              void* d_out, int out_size, void* d_ws, size_t ws_size,
                              hipStream_t stream) {
  (void)in_sizes; (void)n_in; (void)out_size; (void)ws_size;
  constexpr int Bb = 8, L = 1024, E = 768, H = 12;
  constexpr int M = Bb * L;  // 8192 token rows

  const float* x     = (const float*)d_in[0];
  const float* ln1_s = (const float*)d_in[1];
  const float* ln1_b = (const float*)d_in[2];
  const float* w_qkv = (const float*)d_in[3];
  const float* b_qkv = (const float*)d_in[4];
  const float* w_prj = (const float*)d_in[5];
  const float* b_prj = (const float*)d_in[6];
  const float* ln2_s = (const float*)d_in[7];
  const float* ln2_b = (const float*)d_in[8];
  const float* w_fc  = (const float*)d_in[9];
  const float* b_fc  = (const float*)d_in[10];
  const float* w_out = (const float*)d_in[11];
  const float* b_out = (const float*)d_in[12];

  char* ws = (char*)d_ws;
  unsigned short* zbuf = (unsigned short*)ws;            // M*E bf16 (LN1, then LN2)
  size_t off = (size_t)M * E * 2;
  unsigned short* qkvb = (unsigned short*)(ws + off);  off += (size_t)M * 3 * E * 2;
  unsigned short* aob  = (unsigned short*)(ws + off);  off += (size_t)M * E * 2;
  float*          x1   = (float*)(ws + off);           off += (size_t)M * E * 4;
  unsigned short* hb   = (unsigned short*)(ws + off);  // M*(4E) bf16

  const dim3 blk(256);
  ln_bf16_kernel<<<M, blk, 0, stream>>>(x, ln1_s, ln1_b, zbuf);
  gemm_bf16_kernel<0><<<dim3(3 * E / 128, M / 128), blk, 0, stream>>>(
      zbuf, w_qkv, b_qkv, nullptr, qkvb, M, 3 * E, E);
  attn_kernel<<<dim3(L / 128, H, Bb), blk, 0, stream>>>(qkvb, aob);
  gemm_bf16_kernel<1><<<dim3(E / 128, M / 128), blk, 0, stream>>>(
      aob, w_prj, b_prj, x, x1, M, E, E);
  ln_bf16_kernel<<<M, blk, 0, stream>>>(x1, ln2_s, ln2_b, zbuf);
  gemm_bf16_kernel<2><<<dim3(4 * E / 128, M / 128), blk, 0, stream>>>(
      zbuf, w_fc, b_fc, nullptr, hb, M, 4 * E, E);
  gemm_bf16_kernel<1><<<dim3(E / 128, M / 128), blk, 0, stream>>>(
      hb, w_out, b_out, x1, (float*)d_out, M, E, 4 * E);
}
